// TestBlockFP8Model_68212670595881
// MI455X (gfx1250) — 8-bit matrix operand order repaired and checked by reading, compiled, not run on hardware
//
#include <hip/hip_runtime.h>
#include <hip/hip_bf16.h>
#include <stdint.h>


typedef int    v16i __attribute__((ext_vector_type(16)));
typedef float  v8f  __attribute__((ext_vector_type(8)));
typedef float  v4f  __attribute__((ext_vector_type(4)));

#define T_DIM   8192
#define H_DIM   4096
#define KTILES  (H_DIM / 128)
#define FP8MAX  448.0f

__device__ __forceinline__ uint32_t f32_to_e4m3(float f) {
  f = fminf(fmaxf(f, -FP8MAX), FP8MAX);
  uint32_t u    = __float_as_uint(f);
  uint32_t sign = (u >> 24) & 0x80u;
  uint32_t a    = u & 0x7FFFFFFFu;
  float    af   = __uint_as_float(a);
  int      e    = (int)(a >> 23) - 127;
  uint32_t r;
  if (e < -6) {
    r = (uint32_t)(int)rintf(af * 512.0f);
  } else {
    uint32_t lsb = (a >> 20) & 1u;
    uint32_t rd  = a + 0x7FFFFu + lsb;
    uint32_t e2  = rd >> 23;
    uint32_t m3  = (rd >> 20) & 7u;
    r = (((e2 - 120u) & 0xFu) << 3) | m3;
  }
  return sign | r;
}

__device__ __forceinline__ uint32_t pack4_e4m3(float f0, float f1, float f2, float f3) {
  return f32_to_e4m3(f0) | (f32_to_e4m3(f1) << 8) |
         (f32_to_e4m3(f2) << 16) | (f32_to_e4m3(f3) << 24);
}

__global__ void __launch_bounds__(512) quant_x_kernel(
    const float* __restrict__ x, uint8_t* __restrict__ xq,
    float* __restrict__ xscale) {
  __shared__ float smax[512];
  __shared__ float sinv[16];

  const int tid  = threadIdx.x;
  const int lane = tid & 31;
  const int grp  = tid >> 5;
  const int rowTile = blockIdx.x;
  const int row  = rowTile * 16 + grp;

  const v4f* xr = (const v4f*)(x + (size_t)row * H_DIM);
  float m = 0.f;
  #pragma unroll
  for (int i = 0; i < H_DIM / 4 / 32; ++i) {
    v4f v = xr[lane + 32 * i];
    m = fmaxf(m, fmaxf(fmaxf(fabsf(v.x), fabsf(v.y)),
                       fmaxf(fabsf(v.z), fabsf(v.w))));
  }
  smax[tid] = m;
  __syncthreads();
  if (lane == 0) {
    float mm = 0.f;
    #pragma unroll
    for (int i = 0; i < 32; ++i) mm = fmaxf(mm, smax[grp * 32 + i]);
    float sc = fmaxf(mm / FP8MAX, 1e-12f);
    xscale[row] = sc;
    sinv[grp]   = 1.0f / sc;
  }
  __syncthreads();

  const int v  = grp;
  const int M  = lane & 15;
  const int ls = lane >> 4;
  const int K  = 64 * (v >> 3) + 16 * ((v & 7) >> 1) + 8 * ls + 4 * (v & 1);
  const float inv = sinv[M];
  const float* xrow = x + (size_t)(rowTile * 16 + M) * H_DIM;
  uint32_t* dst = (uint32_t*)xq + (size_t)rowTile * KTILES * 512
                + (size_t)lane * 16 + v;
  for (int kt = 0; kt < KTILES; ++kt) {
    v4f f = *(const v4f*)(xrow + kt * 128 + K);
    dst[(size_t)kt * 512] = pack4_e4m3(f.x * inv, f.y * inv, f.z * inv, f.w * inv);
  }
}

__global__ void __launch_bounds__(256) quant_w_kernel(
    const float* __restrict__ w, uint8_t* __restrict__ wq) {
  const size_t d  = (size_t)blockIdx.x * 256 + threadIdx.x;
  const int wd    = (int)(d & 511);
  const int tile  = (int)(d >> 9);
  const int kt    = tile & (KTILES - 1);
  const int nt    = tile / KTILES;
  const int lane  = wd >> 4;
  const int v     = wd & 15;
  const int N     = lane & 15;
  const int ls    = lane >> 4;
  const int Kb    = 64 * (v >> 3) + 16 * ((v & 7) >> 1) + 8 * ls + 4 * (v & 1);
  const float* s  = w + (size_t)(kt * 128 + Kb) * H_DIM + nt * 16 + N;
  ((uint32_t*)wq)[d] =
      pack4_e4m3(s[0], s[H_DIM], s[2 * (size_t)H_DIM], s[3 * (size_t)H_DIM]);
}

__device__ __forceinline__ void stage_tiles_async(
    const uint8_t* __restrict__ xq, const uint8_t* __restrict__ wq,
    uint8_t* smemBuf, int bm, int bn, int kt, int tid) {
  #pragma unroll
  for (int i = 0; i < 8; ++i) {
    const int q    = tid + i * 256;
    const int tile = q >> 7;
    const int off  = (q & 127) * 16;
    const uint8_t* g = (tile < 8)
        ? xq + ((size_t)(bm * 8 + tile) * KTILES + kt) * 2048 + off
        : wq + ((size_t)(bn * 8 + tile - 8) * KTILES + kt) * 2048 + off;
    const uint32_t lds = (uint32_t)(uintptr_t)(smemBuf + tile * 2048 + off);
    const uint64_t ga  = (uint64_t)(uintptr_t)g;
    asm volatile("global_load_async_to_lds_b128 %0, %1, off"
                 :: "v"(lds), "v"(ga) : "memory");
  }
}

__global__ void __launch_bounds__(256, 1) fp8_gemm_kernel(
    const uint8_t* __restrict__ xq, const uint8_t* __restrict__ wq,
    const float* __restrict__ xscale, const float* __restrict__ wscale,
    float* __restrict__ out) {
  __shared__ uint8_t smem[2][32768];

  const int tid  = threadIdx.x;
  const int lane = tid & 31;
  const int wave = tid >> 5;
  const int ws_m = wave >> 2;
  const int ws_n = wave & 3;
  const int bm   = blockIdx.x >> 5;
  const int bn   = blockIdx.x & 31;

  v8f c[4][2];
  #pragma unroll
  for (int i = 0; i < 4; ++i)
    #pragma unroll
    for (int j = 0; j < 2; ++j)
      c[i][j] = (v8f){0.f, 0.f, 0.f, 0.f, 0.f, 0.f, 0.f, 0.f};

  stage_tiles_async(xq, wq, smem[0], bm, bn, 0, tid);
  asm volatile("s_wait_asynccnt 0x0" ::: "memory");
  __syncthreads();

  int buf = 0;
  for (int kt = 0; kt < KTILES; ++kt) {
    if (kt + 1 < KTILES)
      stage_tiles_async(xq, wq, smem[buf ^ 1], bm, bn, kt + 1, tid);

    const uint8_t* sbuf = smem[buf];
    v16i a[4], b[2];
    #pragma unroll
    for (int i = 0; i < 4; ++i)
      a[i] = *(const v16i*)(sbuf + (ws_m * 4 + i) * 2048 + lane * 64);
    #pragma unroll
    for (int j = 0; j < 2; ++j)
      b[j] = *(const v16i*)(sbuf + 16384 + (ws_n * 2 + j) * 2048 + lane * 64);

    #pragma unroll
    for (int i = 0; i < 4; ++i)
      #pragma unroll
      for (int j = 0; j < 2; ++j)
        c[i][j] = __builtin_amdgcn_wmma_f32_16x16x128_fp8_fp8(
            a[i], b[j], (short)0, c[i][j], false, false);

    asm volatile("s_wait_asynccnt 0x0" ::: "memory");
    __syncthreads();
    buf ^= 1;
  }

  const float wsc = wscale[0];
  const int ls  = lane >> 4;
  const int nlo = lane & 15;
  #pragma unroll
  for (int i = 0; i < 4; ++i) {
    const int m0 = bm * 128 + ws_m * 64 + i * 16;
    #pragma unroll
    for (int j = 0; j < 2; ++j) {
      const int n0 = bn * 128 + ws_n * 32 + j * 16;
      #pragma unroll
      for (int r = 0; r < 8; ++r) {
        const int row = m0 + r + 8 * ls;
        out[(size_t)row * H_DIM + n0 + nlo] = c[i][j][r] * xscale[row] * wsc;
      }
    }
  }
}

extern "C" void kernel_launch(void* const* d_in, const int* in_sizes, int n_in,
                              void* d_out, int out_size, void* d_ws, size_t ws_size,
                              hipStream_t stream) {
  (void)in_sizes; (void)n_in; (void)out_size; (void)ws_size;
  const float* x      = (const float*)d_in[0];
  const float* w      = (const float*)d_in[1];
  const float* wscale = (const float*)d_in[2];
  float* out          = (float*)d_out;

  uint8_t* ws = (uint8_t*)d_ws;
  uint8_t* xq = ws;
  uint8_t* wq = ws + (size_t)T_DIM * H_DIM;
  float* xsc  = (float*)(ws + (size_t)T_DIM * H_DIM + (size_t)H_DIM * H_DIM);

  quant_x_kernel<<<T_DIM / 16, 512, 0, stream>>>(x, xq, xsc);
  quant_w_kernel<<<((size_t)H_DIM * H_DIM / 4) / 256, 256, 0, stream>>>(w, wq);
  fp8_gemm_kernel<<<(T_DIM / 128) * (H_DIM / 128), 256, 0, stream>>>(
      xq, wq, xsc, wscale, out);
}
